// AttentionLayer_20761871909361
// MI455X (gfx1250) — compile-verified, matrix operand order checked by reading, not run on hardware
//
#include <hip/hip_runtime.h>
#include <stdint.h>

typedef __bf16 bf16_t;
typedef __attribute__((ext_vector_type(16))) __bf16 v16bf;
typedef __attribute__((ext_vector_type(8)))  float  v8f;

// ---------- helpers ----------
__device__ inline __bf16 f2bf(float x) {
    union { float f; uint32_t u; } c; c.f = x;
    uint32_t r = (c.u + 0x7FFFu + ((c.u >> 16) & 1u)) >> 16;   // RNE truncate
    union { unsigned short s; __bf16 b; } o; o.s = (unsigned short)r;
    return o.b;
}

// 16-bit WMMA fragment layout (ISA 7.12.2): lane m and m+16 hold row m;
// per 32-K chunk: lanes 0-15 get K [0..7]&[16..23], lanes 16-31 get [8..15]&[24..31].
// => per lane: two contiguous 16B segments at +8*kh and +16+8*kh (kh = lane>>4).
__device__ inline v16bf load_frag(const bf16_t* rowk, int kh) {
    union { uint4 u[2]; v16bf v; } f;
    f.u[0] = *(const uint4*)(rowk + 8 * kh);
    f.u[1] = *(const uint4*)(rowk + 16 + 8 * kh);
    return f.v;
}
__device__ inline v16bf load_frag(const float* rowk, int kh) {   // f32 -> bf16 on the fly
    union { __bf16 h[16]; v16bf v; } f;
    const float* p0 = rowk + 8 * kh;
    const float* p1 = rowk + 16 + 8 * kh;
#pragma unroll
    for (int i = 0; i < 8; ++i) { f.h[i] = f2bf(p0[i]); f.h[8 + i] = f2bf(p1[i]); }
    return f.v;
}

// ---------- phase 1: prep ----------
__global__ void k_transpose_cvt(const float* __restrict__ src, bf16_t* __restrict__ dst,
                                int R, int C) {   // src[R][C] f32 -> dst[C][R] bf16
    int i = blockIdx.x * blockDim.x + threadIdx.x;
    if (i < R * C) {
        int r = i / C, c = i - r * C;
        dst[(size_t)c * R + r] = f2bf(src[i]);
    }
}

__global__ void k_feat(const float* __restrict__ ft, float2* __restrict__ ftv, int n) {
    int i = blockIdx.x * blockDim.x + threadIdx.x;
    if (i < n) {
        float v = ft[i];
        bool bad = (v != v);                 // isnan
        ftv[i] = make_float2(bad ? 0.f : v, bad ? 0.f : 1.f);
    }
}

// C[M][N] (bf16) = A[M][K] @ Bt[N][K]^T ; M,N multiples of 16, K multiple of 32.
template <typename AT>
__global__ void k_gemm(const AT* __restrict__ A, const bf16_t* __restrict__ Bt,
                       bf16_t* __restrict__ C, int M, int N, int K) {
    int lane = threadIdx.x & 31;
    int wave = threadIdx.x >> 5;
    int lr = lane & 15, kh = lane >> 4;
    int tilesN = N >> 4;
    int tile = blockIdx.x * (blockDim.x >> 5) + wave;
    if (tile >= (M >> 4) * tilesN) return;
    int tm = tile / tilesN, tn = tile - tm * tilesN;

    const AT*     arow = A  + (size_t)(tm * 16 + lr) * K;
    const bf16_t* brow = Bt + (size_t)(tn * 16 + lr) * K;

    v8f acc = {0.f, 0.f, 0.f, 0.f, 0.f, 0.f, 0.f, 0.f};
    for (int c = 0; c < K; c += 32) {
        v16bf a = load_frag(arow + c, kh);
        v16bf b = load_frag(brow + c, kh);
        acc = __builtin_amdgcn_wmma_f32_16x16x32_bf16(false, a, false, b,
                                                      (short)0, acc, false, false);
    }
    int col = tn * 16 + lr;
#pragma unroll
    for (int v = 0; v < 8; ++v) {              // C/D layout: row = v + 8*kh, col = lr
        int row = tm * 16 + v + 8 * kh;
        C[(size_t)row * N + col] = f2bf(acc[v]);
    }
}

// ---------- phase 2: fused scores + mask + softmax + weighted sum ----------
__global__ __launch_bounds__(256)
void k_attn(const float* __restrict__ bias,     // [ND][NS]
            const bf16_t* __restrict__ q,       // [ND][128] bf16
            const bf16_t* __restrict__ h1,      // [NS][128] bf16
            const float2* __restrict__ ftv,     // [NS] {ft, valid}
            float* __restrict__ out)            // [ND]
{
    constexpr int ND = 10000, NS = 8000, H = 128;
    // exp(s/sqrt(128)) = exp2(s * C_POS);  LeakyReLU folds into selected multiplier
    constexpr float C_POS = 0.12752462158157460f;        // log2(e)/sqrt(128)
    constexpr float C_NEG = 0.02550492431631492f;        // 0.2 * C_POS

    __shared__ bf16_t Bsh[2][64][136];   // double-buffered; +16B row pad -> no bank conflicts

    int tid = threadIdx.x;
    int lane = tid & 31, wave = tid >> 5;
    int lr = lane & 15, kh = lane >> 4;
    int r0 = blockIdx.x * 128 + wave * 16;
    bool waveActive = (r0 < ND);          // ND%16==0 -> tiles are fully in or fully out

    // A fragments (16 q-rows, K=128) live in VGPRs for the whole kernel
    int rowA = waveActive ? (r0 + lr) : 0;
    const bf16_t* arow = q + (size_t)rowA * H;
    v16bf a0 = load_frag(arow + 0,  kh);
    v16bf a1 = load_frag(arow + 32, kh);
    v16bf a2 = load_frag(arow + 64, kh);
    v16bf a3 = load_frag(arow + 96, kh);

    // uniform, always-in-bounds bias base row for this wave's lane-half
    const float* bias_base = bias + (size_t)(waveActive ? (r0 + 8 * kh) : 0) * NS;

    float lsum[8], facc[8];
#pragma unroll
    for (int v = 0; v < 8; ++v) { lsum[v] = 0.f; facc[v] = 0.f; }

    // ---- async DMA staging: global -> LDS, no VGPR staging, tracked by ASYNCcnt ----
    int srow = tid >> 4, sseg = tid & 15;       // this thread's 16B segment (rows step by 16)
    auto stage_async = [&](int j0, int buf) {
#pragma unroll
        for (int s = 0; s < 4; ++s) {
            int row = srow + 16 * s;
            uint32_t lds = (uint32_t)(uintptr_t)&Bsh[buf][row][sseg * 8];
            uint64_t ga  = (uint64_t)(uintptr_t)(h1 + (size_t)(j0 + row) * H + sseg * 8);
            asm volatile("global_load_async_to_lds_b128 %0, %1, off"
                         :: "v"(lds), "v"(ga) : "memory");
        }
    };

    stage_async(0, 0);
    asm volatile("s_wait_asynccnt 0x0" ::: "memory");
    __syncthreads();

    for (int j0 = 0, it = 0; j0 < NS; j0 += 64, ++it) {
        int buf = it & 1;
        if (j0 + 64 < NS) stage_async(j0 + 64, buf ^ 1);   // DMA next tile during compute

#pragma unroll
        for (int t = 0; t < 4; ++t) {
            int col = j0 + t * 16 + lr;

            // issue the 8 streaming bias loads first (latency hidden by WMMA chain)
            const float* bp = bias_base + col;
            float bv[8];
#pragma unroll
            for (int v = 0; v < 8; ++v)
                bv[v] = __builtin_nontemporal_load(bp + (size_t)v * NS);
            float2 fv = ftv[col];

            int rb = t * 16 + lr;                 // B lane = column of e = row of h1
            union { uint4 u[2]; v16bf v; } b0, b1, b2, b3;
            b0.u[0] = *(const uint4*)&Bsh[buf][rb][0   + 8 * kh];
            b0.u[1] = *(const uint4*)&Bsh[buf][rb][16  + 8 * kh];
            b1.u[0] = *(const uint4*)&Bsh[buf][rb][32  + 8 * kh];
            b1.u[1] = *(const uint4*)&Bsh[buf][rb][48  + 8 * kh];
            b2.u[0] = *(const uint4*)&Bsh[buf][rb][64  + 8 * kh];
            b2.u[1] = *(const uint4*)&Bsh[buf][rb][80  + 8 * kh];
            b3.u[0] = *(const uint4*)&Bsh[buf][rb][96  + 8 * kh];
            b3.u[1] = *(const uint4*)&Bsh[buf][rb][112 + 8 * kh];

            v8f acc = {0.f, 0.f, 0.f, 0.f, 0.f, 0.f, 0.f, 0.f};
            acc = __builtin_amdgcn_wmma_f32_16x16x32_bf16(false, a0, false, b0.v, (short)0, acc, false, false);
            acc = __builtin_amdgcn_wmma_f32_16x16x32_bf16(false, a1, false, b1.v, (short)0, acc, false, false);
            acc = __builtin_amdgcn_wmma_f32_16x16x32_bf16(false, a2, false, b2.v, (short)0, acc, false, false);
            acc = __builtin_amdgcn_wmma_f32_16x16x32_bf16(false, a3, false, b3.v, (short)0, acc, false, false);

#pragma unroll
            for (int v = 0; v < 8; ++v) {         // fully branchless per-element math
                float s = acc[v];
                float m = (s > 0.f) ? C_POS : C_NEG;          // LeakyReLU + scale + ln2
                float p = __builtin_amdgcn_exp2f(s * m);      // v_exp_f32
                float g = (bv[v] > 0.f) ? fv.y : 0.f;         // adjacency & NaN masks
                p *= g;                                        // masked -> 0
                lsum[v] += p;
                facc[v] += p * fv.x;
            }
        }

        asm volatile("s_wait_asynccnt 0x0" ::: "memory");     // next tile landed in LDS
        __syncthreads();
    }

    // reduce numerator/denominator across the 16 lanes holding each row, then store
#pragma unroll
    for (int v = 0; v < 8; ++v) {
        float l = lsum[v], f = facc[v];
#pragma unroll
        for (int off = 1; off < 16; off <<= 1) {
            l += __shfl_xor(l, off, 32);
            f += __shfl_xor(f, off, 32);
        }
        if (lr == v && waveActive) out[r0 + v + 8 * kh] = f / l;
    }
}

// ---------- launch ----------
extern "C" void kernel_launch(void* const* d_in, const int* in_sizes, int n_in,
                              void* d_out, int out_size, void* d_ws, size_t ws_size,
                              hipStream_t stream) {
    const float* bias     = (const float*)d_in[0];
    const float* emb_dest = (const float*)d_in[1];
    const float* emb_src  = (const float*)d_in[2];
    const float* ft       = (const float*)d_in[3];
    const float* W        = (const float*)d_in[4];
    const float* W2       = (const float*)d_in[5];
    float* out = (float*)d_out;

    constexpr int ND = 10000, NS = 8000, IN = 256, H = 128;

    char* ws = (char*)d_ws;
    size_t off = 0;
    auto alloc = [&](size_t bytes) {
        void* p = ws + off;
        off = (off + bytes + 255) & ~(size_t)255;
        return p;
    };
    bf16_t* q_bf  = (bf16_t*)alloc((size_t)ND * H * 2);
    bf16_t* h1_bf = (bf16_t*)alloc((size_t)NS * H * 2);
    bf16_t* h2_bf = (bf16_t*)alloc((size_t)ND * H * 2);
    bf16_t* Wt    = (bf16_t*)alloc((size_t)IN * H * 2);    // W^T  [128][256]
    bf16_t* W2t   = (bf16_t*)alloc((size_t)H * H * 2);     // W2^T [128][128]
    float2* ftv   = (float2*)alloc((size_t)NS * sizeof(float2));
    (void)ws_size; (void)in_sizes; (void)n_in; (void)out_size;

    k_transpose_cvt<<<(IN * H + 255) / 256, 256, 0, stream>>>(W,  Wt,  IN, H);
    k_transpose_cvt<<<(H * H  + 255) / 256, 256, 0, stream>>>(W2, W2t, H, H);
    k_feat<<<(NS + 255) / 256, 256, 0, stream>>>(ft, ftv, NS);

    // h2 = emb_dest @ W ; h1 = emb_src @ W ; q = h2 @ W2   (A is f32 for the first two)
    {
        int tiles = (ND / 16) * (H / 16);
        k_gemm<float><<<(tiles + 3) / 4, 128, 0, stream>>>(emb_dest, Wt, h2_bf, ND, H, IN);
    }
    {
        int tiles = (NS / 16) * (H / 16);
        k_gemm<float><<<(tiles + 3) / 4, 128, 0, stream>>>(emb_src, Wt, h1_bf, NS, H, IN);
    }
    {
        int tiles = (ND / 16) * (H / 16);
        k_gemm<bf16_t><<<(tiles + 3) / 4, 128, 0, stream>>>(h2_bf, W2t, q_bf, ND, H, H);
    }

    k_attn<<<(ND + 127) / 128, 256, 0, stream>>>(bias, q_bf, h1_bf, ftv, out);
}
